// HierDDLts_85744727097703
// MI455X (gfx1250) — matrix operand order repaired and checked by reading, not compiled, not run on hardware
//
#include <hip/hip_runtime.h>
#include <math.h>

// ---------------------------------------------------------------------------
// CDNA5 (gfx1250) implementation. wave32.
//  - Matrix ops: __builtin_amdgcn_wmma_f32_16x16x32_f16 (f16 in, f32 acc)
//  - Async data movement: global_load_async_to_lds_b128 (ASYNCcnt) into
//    double-buffered LDS for the sequence-contraction GEMM.
// Problem constants: B=16, S=1024, Din=D=96, G=8, T1=1024, T2=512.
// ---------------------------------------------------------------------------

typedef __attribute__((ext_vector_type(16))) _Float16 v16h;  // 8 VGPRs
typedef __attribute__((ext_vector_type(8)))  float    v8f;   // 8 VGPRs
typedef __attribute__((ext_vector_type(4)))  float    f32x4; // trivial 16B POD

union Frag {
    v16h  v;
    f32x4 q[2];
};

constexpr int Bc = 16;    // batch
constexpr int Sc = 1024;  // sequence length (K of the big GEMM)
constexpr int Dc = 96;    // feature dim

// WMMA fragment layout assumptions (CDNA5 ISA 7.12.2, wave32):
//  A (16x32 f16): lane L<16 : row M=L,   K = {0..7 in v0-3, 16..23 in v4-7}
//                 lane L>=16: row M=L-16, K = {8..15, 24..31}
//  B (32x16 f16): lane L<16 : col N=L,   K = {0..7 in v0-3, 16..23 in v4-7}
//                 lane L>=16: col N=L-16, K = {8..15, 24..31}
//  C/D (16x16 f32): vgpr r, lane L<16: (M=r, N=L); lane L>=16: (M=r+8, N=L-16)

__device__ __forceinline__ void wait_async0() {
#if __has_builtin(__builtin_amdgcn_s_wait_asynccnt)
    __builtin_amdgcn_s_wait_asynccnt(0);
#else
    asm volatile("s_wait_asynccnt 0" ::: "memory");
#endif
}

// Async copy 16B/lane from global to LDS (tracked by ASYNCcnt).
// lds_off = LDS byte offset (low 32 bits of a generic pointer to __shared__).
__device__ __forceinline__ void async_ld_b128(unsigned int lds_off, const void* g) {
    asm volatile("global_load_async_to_lds_b128 %0, %1, off"
                 :: "v"(lds_off), "v"((unsigned long long)(uintptr_t)g)
                 : "memory");
}

// ---------------------------------------------------------------------------
// Kernel 1: Z0 = X @ M^T ; optional LayerNorm; f16 output row-major or
// transposed (B,D,S). One block per (b,s) row, 128 threads.
// ---------------------------------------------------------------------------
template <bool LN, bool TRANSPOSED, bool STORE_Z0>
__global__ __launch_bounds__(128) void linproj_kernel(
    const float* __restrict__ X,     // (B,S,96)
    const float* __restrict__ M,     // (96,96) row-major M[d][k]
    const float* __restrict__ lnw,
    const float* __restrict__ lnb,
    float*       __restrict__ Z0out, // (B,S,96) fp32
    _Float16*    __restrict__ Hout)  // f16 output
{
    __shared__ float xr[Dc];
    __shared__ float zr[Dc];
    const int bs = blockIdx.x;           // b*S + s
    const int b  = bs >> 10;
    const int s  = bs & (Sc - 1);
    const int d  = threadIdx.x;

    if (d < Dc) xr[d] = X[(size_t)bs * Dc + d];
    __syncthreads();

    if (d < Dc) {
        const float* mrow = M + d * Dc;
        float z = 0.f;
        #pragma unroll 8
        for (int k = 0; k < Dc; ++k) z += xr[k] * mrow[k];
        zr[d] = z;
        if (STORE_Z0) Z0out[(size_t)bs * Dc + d] = z;
    }
    __syncthreads();

    if (d < Dc) {
        float val = zr[d];
        if (LN) {
            float mu = 0.f;
            for (int k = 0; k < Dc; ++k) mu += zr[k];
            mu *= (1.0f / Dc);
            float var = 0.f;
            for (int k = 0; k < Dc; ++k) { float t = zr[k] - mu; var += t * t; }
            var *= (1.0f / Dc);
            val = (val - mu) * rsqrtf(var + 1e-5f) * lnw[d] + lnb[d];
        }
        if (TRANSPOSED) Hout[((size_t)b * Dc + d) * Sc + s] = (_Float16)val;
        else            Hout[(size_t)bs * Dc + d]           = (_Float16)val;
    }
}

// ---------------------------------------------------------------------------
// Kernel 2: fp32 (R,C) -> f16 (C,R) transpose (for Linker matrices).
// ---------------------------------------------------------------------------
__global__ __launch_bounds__(256) void transpose_half_kernel(
    const float* __restrict__ In, _Float16* __restrict__ Outp, int R, int C)
{
    int idx = blockIdx.x * blockDim.x + threadIdx.x;
    if (idx >= R * C) return;
    int r = idx / C, c = idx % C;
    Outp[(size_t)c * R + r] = (_Float16)In[idx];
}

// ---------------------------------------------------------------------------
// Kernel 3: per-position transform.
//   W[s][i][j] = sum_g P[i][j][g] * cos(2*pi*s / (i*768 + j*8 + g + 2))
//   T[b,s,i]   = sum_j Z[b,s,j] * W[s][i][j]   (16x96x96 GEMM, M=batch)
//   A[b,s,i]   = T (+ Z0 shared residual, layer 1)
// Output transposed: At[(b*96+i)*S + s] -> WMMA-B operand of sequence GEMM.
// One block per position s; 6 waves; W built in LDS; 3 WMMAs per wave.
// ---------------------------------------------------------------------------
template <bool RESID>
__global__ __launch_bounds__(192) void posmix_kernel(
    const _Float16* __restrict__ Zh,   // (B,S,96) f16, layernormed
    const float*    __restrict__ Z0,   // (B,S,96) fp32 residual
    const float*    __restrict__ P,    // (96,96,8)
    _Float16*       __restrict__ Atout)// (B,96,S) f16
{
    __shared__ _Float16 Wsh[Dc * 104];   // padded rows: 104 halves = 208B
    const int s   = blockIdx.x;
    const int tid = threadIdx.x;
    const float twopi_s = 6.28318530717958647692f * (float)s;

    // --- build W[s] in LDS (48 entries/thread, uniform trip count) ---
    for (int e = tid; e < Dc * Dc; e += 192) {
        const int i = e / Dc, j = e - i * Dc;
        const int perbase = i * (Dc * 8) + j * 8 + 2;
        const float* prow = P + (size_t)e * 8;
        float accw = 0.f;
        #pragma unroll
        for (int g = 0; g < 8; ++g)
            accw += prow[g] * cosf(twopi_s / (float)(perbase + g));
        Wsh[i * 104 + j] = (_Float16)accw;
    }
    __syncthreads();

    // --- WMMA: C(16x16) = Z(16x96) x W-tile ---
    const int lane = tid & 31;
    const int wave = tid >> 5;            // 0..5 = N-tile
    const int n0   = wave * 16;
    const int mr   = lane & 15;
    const int ka   = (lane & 16) ? 8 : 0;
    const int kb   = (lane & 16) ? 8 : 0;
    const _Float16* arow = Zh + ((size_t)mr * Sc + s) * Dc;

    v8f acc = {};
    #pragma unroll
    for (int k0 = 0; k0 < Dc; k0 += 32) {
        Frag fa;
        fa.q[0] = *reinterpret_cast<const f32x4*>(arow + k0 + ka);
        fa.q[1] = *reinterpret_cast<const f32x4*>(arow + k0 + ka + 16);
        Frag fb;   // B[k][n] = W[n][k]
        const _Float16* brow = &Wsh[(n0 + mr) * 104 + k0 + kb];
        fb.q[0] = *reinterpret_cast<const f32x4*>(brow);
        fb.q[1] = *reinterpret_cast<const f32x4*>(brow + 16);
        acc = __builtin_amdgcn_wmma_f32_16x16x32_f16(
                  false, fa.v, false, fb.v, (short)0, acc, false, false);
    }

    // --- store (+ residual), transposed (B,96,S); constant strides ---
    const int rlo = (lane & 16) ? 8 : 0;
    const int ii  = n0 + mr;
    _Float16*    obase = Atout + (size_t)ii * Sc + s + (size_t)rlo * (Dc * Sc);
    const float* rbase = Z0 + ((size_t)rlo * Sc + s) * Dc + ii;
    #pragma unroll
    for (int r = 0; r < 8; ++r) {
        float v = acc[r];
        if (RESID) v += rbase[(size_t)r * (Sc * Dc)];
        obase[(size_t)r * (Dc * Sc)] = (_Float16)v;
    }
}

// ---------------------------------------------------------------------------
// Kernel 4: sequence contraction  Out[b,t,d] (+)= sum_s Lt[t,s] * At[b,d,s].
// 8 waves/block; wave owns a 16-row M-tile and 6 N-tiles (48 VGPR acc).
// B-operand K-chunks (96 rows x 32 halves) staged by async copies into
// double-buffered LDS (rows padded to 40 halves for bank spread); waves
// 0..5 issue the async loads; one s_wait_asynccnt+barrier per K-step.
// ---------------------------------------------------------------------------
template <int T, bool ACC>
__global__ __launch_bounds__(256) void gemm_seq_kernel(
    const _Float16* __restrict__ Lt,  // (T, 1024) f16
    const _Float16* __restrict__ At,  // (B, 96, 1024) f16
    float*          __restrict__ Out) // (B, T, 96) fp32
{
    __shared__ _Float16 Bsh[2][Dc * 40];   // 2 x 7680 B

    const int tid  = threadIdx.x;
    const int lane = tid & 31;
    const int wave = tid >> 5;
    const int b    = blockIdx.y;
    const int t0   = blockIdx.x * 128 + wave * 16;
    const int ka   = (lane & 16) ? 8 : 0;
    const int kb   = (lane & 16) ? 8 : 0;

    const _Float16* arow  = Lt + (size_t)(t0 + (lane & 15)) * Sc + ka;
    const _Float16* bbase = At + (size_t)b * Dc * Sc;

    // 96 rows * 4 segments of 8 halves = 384 pieces; threads 0..191 x2.
    auto issue = [&](int k0, int bufsel) {
        if (tid < 192) {
            #pragma unroll
            for (int it = 0; it < 2; ++it) {
                const int p   = tid + it * 192;
                const int row = p >> 2;
                const int seg = p & 3;
                const _Float16* g = bbase + (size_t)row * Sc + k0 + seg * 8;
                const unsigned int l =
                    (unsigned int)(uintptr_t)&Bsh[bufsel][row * 40 + seg * 8];
                async_ld_b128(l, g);
            }
        }
    };

    v8f acc[6];
    #pragma unroll
    for (int j = 0; j < 6; ++j) acc[j] = (v8f){};

    issue(0, 0);

    constexpr int NK = Sc / 32;   // 32 K-steps
    for (int kc = 0; kc < NK; ++kc) {
        wait_async0();
        __syncthreads();
        if (kc + 1 < NK) issue((kc + 1) * 32, (kc + 1) & 1);

        const _Float16* bufp = &Bsh[kc & 1][0];
        Frag fa;
        const _Float16* ap = arow + kc * 32;
        fa.q[0] = *reinterpret_cast<const f32x4*>(ap);
        fa.q[1] = *reinterpret_cast<const f32x4*>(ap + 16);
        #pragma unroll
        for (int j = 0; j < 6; ++j) {
            const _Float16* brow = bufp + (j * 16 + (lane & 15)) * 40 + kb;
            Frag fb;
            fb.q[0] = *reinterpret_cast<const f32x4*>(brow);
            fb.q[1] = *reinterpret_cast<const f32x4*>(brow + 16);
            acc[j] = __builtin_amdgcn_wmma_f32_16x16x32_f16(
                         false, fa.v, false, fb.v, (short)0, acc[j],
                         false, false);
        }
        // no trailing barrier needed: buffer (kc&1) is only overwritten by
        // issue(kc+2), which every wave reaches only after the barrier at
        // iteration kc+1, by which time this iteration's LDS reads are done.
    }

    const int rlo = (lane & 16) ? 8 : 0;
    const int c   = lane & 15;
    float* obase = Out + ((size_t)b * T + t0 + rlo) * Dc + c;
    #pragma unroll
    for (int j = 0; j < 6; ++j) {
        #pragma unroll
        for (int r = 0; r < 8; ++r) {
            float* o = obase + (size_t)r * Dc + j * 16;
            const float v = acc[j][r];
            *o = ACC ? (*o + v) : v;
        }
    }
}

// ---------------------------------------------------------------------------
// Launcher
// ---------------------------------------------------------------------------
extern "C" void kernel_launch(void* const* d_in, const int* in_sizes, int n_in,
                              void* d_out, int out_size, void* d_ws, size_t ws_size,
                              hipStream_t stream) {
    const float* x    = (const float*)d_in[0];
    const float* M1   = (const float*)d_in[1];
    const float* P1   = (const float*)d_in[2];
    const float* L1   = (const float*)d_in[3];
    const float* ln1w = (const float*)d_in[4];
    const float* ln1b = (const float*)d_in[5];
    const float* M2   = (const float*)d_in[6];
    const float* P2   = (const float*)d_in[7];
    const float* L2   = (const float*)d_in[8];
    const float* ln2w = (const float*)d_in[9];
    const float* ln2b = (const float*)d_in[10];
    const float* Rw2  = (const float*)d_in[11];
    const float* Rl2  = (const float*)d_in[12];
    float* out = (float*)d_out;

    const int B = Bc, S = Sc, D = Dc, T1 = 1024, T2 = 512;
    (void)in_sizes; (void)n_in; (void)out_size; (void)ws_size;

    char* w = (char*)d_ws;
    auto carve = [&](size_t bytes) -> char* {
        char* p = w;
        w += (bytes + 255) & ~(size_t)255;
        return p;
    };
    _Float16* L1t  = (_Float16*)carve((size_t)T1 * S * sizeof(_Float16));
    _Float16* L2t  = (_Float16*)carve((size_t)T2 * S * sizeof(_Float16));
    _Float16* Rl2t = (_Float16*)carve((size_t)T2 * S * sizeof(_Float16));
    float*    Z0_1 = (float*)   carve((size_t)B * S * D * sizeof(float));
    _Float16* Zh1  = (_Float16*)carve((size_t)B * S * D * sizeof(_Float16));
    _Float16* A1t  = (_Float16*)carve((size_t)B * D * S * sizeof(_Float16));
    float*    h    = (float*)   carve((size_t)B * T1 * D * sizeof(float));
    _Float16* Zh2  = (_Float16*)carve((size_t)B * S * D * sizeof(_Float16));
    _Float16* Rt   = (_Float16*)carve((size_t)B * D * S * sizeof(_Float16));
    _Float16* A2t  = (_Float16*)carve((size_t)B * D * S * sizeof(_Float16));

    // ---- parameter transposes (f32 -> f16, (S,T) -> (T,S)) ----
    {
        int n = S * T1;
        transpose_half_kernel<<<(n + 255) / 256, 256, 0, stream>>>(L1, L1t, S, T1);
        n = S * T2;
        transpose_half_kernel<<<(n + 255) / 256, 256, 0, stream>>>(L2, L2t, S, T2);
        transpose_half_kernel<<<(n + 255) / 256, 256, 0, stream>>>(Rl2, Rl2t, S, T2);
    }

    // ---- layer 1 ----
    linproj_kernel<true, false, true><<<B * S, 128, 0, stream>>>(
        x, M1, ln1w, ln1b, Z0_1, Zh1);
    posmix_kernel<true><<<S, 192, 0, stream>>>(Zh1, Z0_1, P1, A1t);
    gemm_seq_kernel<1024, false><<<dim3(T1 / 128, B), 256, 0, stream>>>(L1t, A1t, h);

    // ---- layer 2 ----
    linproj_kernel<true, false, false><<<B * S, 128, 0, stream>>>(
        h, M2, ln2w, ln2b, nullptr, Zh2);
    linproj_kernel<false, true, false><<<B * S, 128, 0, stream>>>(
        h, Rw2, ln2w, ln2b, nullptr, Rt);
    posmix_kernel<false><<<S, 192, 0, stream>>>(Zh2, nullptr, P2, A2t);
    gemm_seq_kernel<512, false><<<dim3(T2 / 128, B), 256, 0, stream>>>(L2t, A2t, out);
    gemm_seq_kernel<512, true><<<dim3(T2 / 128, B), 256, 0, stream>>>(Rl2t, Rt, out);
}
